// MLAPlusMoEBlock_29721173688615
// MI455X (gfx1250) — matrix operand order repaired and checked by reading, not compiled, not run on hardware
//
#include <hip/hip_runtime.h>
#include <hip/hip_bf16.h>
#include <math.h>
#include <stdint.h>

// ---------------------------------------------------------------------------
// MLA + MoE block for MI455X (gfx1250, wave32, WMMA + Tensor Data Mover).
// All GEMMs run through v_wmma_f32_16x16x32_bf16; the A tile is staged into
// LDS by the TDM (tensor_load_to_lds + s_wait_tensorcnt), using the TDM's
// LDS-padding feature to produce the bank-conflict-avoiding row pitch.
// Every matrix dim in this problem is a multiple of the 128x128x32 tile.
// ---------------------------------------------------------------------------

typedef __attribute__((ext_vector_type(16))) __bf16 v16bf;
typedef __attribute__((ext_vector_type(8)))  float  v8f;
typedef __attribute__((ext_vector_type(4)))  unsigned int u32x4;
typedef __attribute__((ext_vector_type(8)))  int i32x8;
typedef __attribute__((ext_vector_type(4)))  int i32x4;

#define S_  512
#define B_  4
#define D_  2048
#define H_  16
#define HD_ 128
#define L_  512
#define E_  8
#define F_  1024

// TDM builtin arity differs between toolchains (probe doc): ROCm 7.2/clang-22
// takes 5 args, amdgpu-toolchain/clang-23 takes 6 (extra i32x8 group).
#if defined(__clang_major__) && (__clang_major__ >= 23)
#define TDM_LOAD(g0, g1, g2, g3) \
  __builtin_amdgcn_tensor_load_to_lds((g0), (g1), (g2), (g3), (i32x8){0,0,0,0,0,0,0,0}, 0)
#else
#define TDM_LOAD(g0, g1, g2, g3) \
  __builtin_amdgcn_tensor_load_to_lds((g0), (g1), (g2), (g3), 0)
#endif

// ---------------------------------------------------------------------------
// Batched strided GEMM:  C[z] = alpha * A[z] (MxK, bf16) * op(B[z]) + epilogue
//   TRANSB=0: B is K x N row-major; TRANSB=1: B is N x K row-major (A * B^T)
// Epilogue: v = alpha*acc; if rowScale: v *= rowScale[row*rsStride];
//           if Cin: v += Cin[row,col];   Cout[row,col] = v
// Block = 256 threads = 8 wave32 waves (2x4); wave owns 64x32 = 4x2 frags.
// LDS layouts: smA[M][K] (pitch 40 via TDM padding), smB[N][K] (pitch 40) so
// both A and B fragments are two 16B runs per lane -> ds_load_b128 pairs.
// ---------------------------------------------------------------------------
template <bool TRANSB>
__global__ __launch_bounds__(256) void gemm_bf16_wmma(
    const __bf16* __restrict__ A, const __bf16* __restrict__ B,
    float* __restrict__ Cout, const float* __restrict__ Cin,
    const float* __restrict__ rowScale, int rsStride,
    int M, int N, int K, int lda, int ldb, int ldc,
    long long sA, long long sB, long long sC, float alpha)
{
  constexpr int BM = 128, BN = 128, BK = 32;
  constexpr int LA = BK + 8;   // 40 bf16 = 80B row pitch (multiple of 16B)
  constexpr int LB = BK + 8;   // smB is [BN][BK], same pitch
  __shared__ __bf16 smA[BM * LA];
  __shared__ __bf16 smB[BN * LB];

  const long long z = blockIdx.z;
  A += z * sA;
  B += z * sB;
  Cout += z * sC;
  const float* CinZ = Cin ? (Cin + z * sC) : nullptr;

  const int bm   = blockIdx.y * BM;
  const int bn   = blockIdx.x * BN;
  const int tid  = threadIdx.x;
  const int lane = tid & 31;
  const int wave = tid >> 5;
  const int wm   = wave >> 2;   // 0..1  (64 rows each)
  const int wn   = wave & 3;    // 0..3  (32 cols each)
  const int l16  = lane & 15;
  const int half = lane >> 4;
  const int waveId = __builtin_amdgcn_readfirstlane(wave);  // scalar branch key

  // ---- static parts of the TDM descriptor for the A tile --------------------
  // Group1: data_size=2B (code 1), pad_enable, pad_interval code 3 (16 DWORDs
  // = one 64B row), pad_amount code 3 (4 DWORDs = 16B) -> LDS pitch 80B = LA.
  // tensor dims = tile dims (tiles always in bounds), dim0 stride = lda.
  const unsigned ldsA  = (unsigned)(uintptr_t)&smA[0];
  const unsigned g1w0  = (1u << 16) | (1u << 20) | (3u << 22) | (3u << 25);
  const unsigned long long strideE = (unsigned long long)lda;  // elements
  const i32x8 g1 = {
      (int)g1w0,
      (int)(((unsigned)BK & 0xFFFFu) << 16),          // tensor_dim0[15:0]
      (int)((((unsigned)BK >> 16) & 0xFFFFu) |
            (((unsigned)BM & 0xFFFFu) << 16)),        // tdim0 hi | tensor_dim1 lo
      (int)((((unsigned)BM >> 16) & 0xFFFFu) |
            ((unsigned)BK << 16)),                    // tdim1 hi | tile_dim0=32
      (int)BM,                                        // tile_dim1=128, tile_dim2=0
      (int)(strideE & 0xFFFFFFFFull),                 // tensor_dim0_stride lo
      (int)((strideE >> 32) & 0xFFFFull),             // stride hi | tdim1_stride lo
      0 };
  const i32x4 gz = {0, 0, 0, 0};

  v8f acc[4][2];
#pragma unroll
  for (int i = 0; i < 4; ++i)
#pragma unroll
    for (int j = 0; j < 2; ++j) {
      v8f zv = {0.f, 0.f, 0.f, 0.f, 0.f, 0.f, 0.f, 0.f};
      acc[i][j] = zv;
    }

  union V16 { v16bf v; uint4 q[2]; };

  const int KT = K / BK;
  for (int kt = 0; kt < KT; ++kt) {
    const int k0 = kt * BK;

    // ---- A tile via Tensor Data Mover (wave 0 only; EXEC-independent) ------
    if (waveId == 0) {
      const unsigned long long ga =
          (unsigned long long)(uintptr_t)(A + (long long)bm * lda + k0);
      const u32x4 g0 = {
          1u,                                         // count=1 valid descriptor
          ldsA,                                       // lds_addr
          (unsigned)(ga & 0xFFFFFFFFull),             // global_addr[31:0]
          (unsigned)(((ga >> 32) & 0x01FFFFFFull) | (2u << 30)) };  // addr hi|type=2
      TDM_LOAD(g0, g1, gz, gz);
    }

    // ---- stage B tile into smB[n][k] (all 8 waves) -------------------------
    if (!TRANSB) {
      // global B is KxN row-major: vector load 8 n-contiguous, scatter over n.
#pragma unroll
      for (int i = 0; i < 2; ++i) {
        const int c  = tid + i * 256;
        const int kk = c >> 4;           // K row (32 rows, 16 chunks each)
        const int nc = (c & 15) * 8;
        union { uint4 u; __bf16 e[8]; } v;
        v.u = *(const uint4*)(B + (long long)(k0 + kk) * ldb + bn + nc);
#pragma unroll
        for (int j = 0; j < 8; ++j) smB[(nc + j) * LB + kk] = v.e[j];
      }
    } else {
      // global B is NxK row-major: straight vectorized copy.
#pragma unroll
      for (int i = 0; i < 2; ++i) {
        const int c  = tid + i * 256;
        const int nn = c >> 2;           // N row (128 rows, 4 chunks each)
        const int kc = (c & 3) * 8;
        const uint4 v = *(const uint4*)(B + (long long)(bn + nn) * ldb + k0 + kc);
        *(uint4*)(&smB[nn * LB + kc]) = v;
      }
    }

    // Prefetch next B tile into L2 while this one computes.
    if (kt + 1 < KT) {
      if (!TRANSB)
        __builtin_prefetch(B + (long long)(k0 + BK + (tid >> 4)) * ldb + bn, 0, 0);
      else
        __builtin_prefetch(B + (long long)(bn + (tid >> 1)) * ldb + k0 + BK, 0, 0);
    }

    if (waveId == 0) __builtin_amdgcn_s_wait_tensorcnt(0);
    __syncthreads();

    // ---- fragments: both are two contiguous 16B runs per lane --------------
    // A 16x32 bf16 (ISA 7.12.2): lane half h covers K = h*8+0..7 and 16+h*8+0..7.
    v16bf afr[4];
#pragma unroll
    for (int mt = 0; mt < 4; ++mt) {
      const __bf16* pa = &smA[(wm * 64 + mt * 16 + l16) * LA];
      V16 t;
      t.q[0] = *(const uint4*)(pa + half * 8);
      t.q[1] = *(const uint4*)(pa + 16 + half * 8);
      afr[mt] = t.v;
    }
    // B 32x16 bf16: lane (N=l16, half h) covers K = h*8+0..7 and 16+h*8+0..7.
    v16bf bfr[2];
#pragma unroll
    for (int nt = 0; nt < 2; ++nt) {
      const __bf16* pb = &smB[(wn * 32 + nt * 16 + l16) * LB];
      V16 t;
      t.q[0] = *(const uint4*)(pb + half * 8);
      t.q[1] = *(const uint4*)(pb + 16 + half * 8);
      bfr[nt] = t.v;
    }

#pragma unroll
    for (int mt = 0; mt < 4; ++mt)
#pragma unroll
      for (int nt = 0; nt < 2; ++nt)
        acc[mt][nt] = __builtin_amdgcn_wmma_f32_16x16x32_bf16(
            false, afr[mt], false, bfr[nt], (short)0, acc[mt][nt], false, false);
    __syncthreads();
  }

  // ---- epilogue: C 16x16 f32 layout: VGPR r holds M=r (lanes 0-15) / M=r+8 ----
#pragma unroll
  for (int mt = 0; mt < 4; ++mt) {
#pragma unroll
    for (int nt = 0; nt < 2; ++nt) {
#pragma unroll
      for (int r = 0; r < 8; ++r) {
        const int row = bm + wm * 64 + mt * 16 + r + half * 8;
        const int col = bn + wn * 32 + nt * 16 + l16;
        float v = alpha * acc[mt][nt][r];
        if (rowScale) v *= rowScale[(long long)row * rsStride];
        if (CinZ) v += CinZ[(long long)row * ldc + col];
        Cout[(long long)row * ldc + col] = v;
      }
    }
  }
}

// ---------------------------------------------------------------------------
// Elementwise / reduction kernels
// ---------------------------------------------------------------------------
__global__ void cast_f32_to_bf16(const float* __restrict__ in,
                                 __bf16* __restrict__ out, long long n) {
  for (long long i = (long long)blockIdx.x * blockDim.x + threadIdx.x; i < n;
       i += (long long)gridDim.x * blockDim.x)
    out[i] = (__bf16)in[i];
}

__global__ void zero_f32(float* __restrict__ p, long long n) {
  for (long long i = (long long)blockIdx.x * blockDim.x + threadIdx.x; i < n;
       i += (long long)gridDim.x * blockDim.x)
    p[i] = 0.f;
}

// out row r (token order depends on tOrder) = rmsnorm(x[src,:]) * w, as bf16.
__global__ __launch_bounds__(256) void rmsnorm_to_bf16(
    const float* __restrict__ x, const float* __restrict__ w,
    __bf16* __restrict__ out, int D, int Sdim, int Bdim, int tOrder) {
  const int r = blockIdx.x;
  int src = r;
  if (tOrder) { const int b = r / Sdim; const int s = r - b * Sdim; src = s * Bdim + b; }
  const float* xr = x + (long long)src * D;
  __shared__ float red[256];
  float ss = 0.f;
  for (int d = threadIdx.x; d < D; d += 256) { const float v = xr[d]; ss += v * v; }
  red[threadIdx.x] = ss;
  __syncthreads();
  for (int o = 128; o > 0; o >>= 1) {
    if (threadIdx.x < o) red[threadIdx.x] += red[threadIdx.x + o];
    __syncthreads();
  }
  const float rms = rsqrtf(red[0] / (float)D + 1e-5f);
  __bf16* orow = out + (long long)r * D;
  for (int d = threadIdx.x; d < D; d += 256)
    orow[d] = (__bf16)(xr[d] * rms * w[d]);
}

// RoPE over (S,B,H,HD) fp32 -> bf16, one thread per (s,b,h, i<HD/2).
__global__ void rope_to_bf16(const float* __restrict__ t, __bf16* __restrict__ o,
                             int Sdim, int Bdim, int Hdim, int HDdim) {
  const int halfd = HDdim / 2;
  const long long total = (long long)Sdim * Bdim * Hdim * halfd;
  long long idx = (long long)blockIdx.x * blockDim.x + threadIdx.x;
  if (idx >= total) return;
  const int i = (int)(idx % halfd);
  long long rest = idx / halfd;
  const int h = (int)(rest % Hdim); rest /= Hdim;
  const int b = (int)(rest % Bdim);
  const int s = (int)(rest / Bdim);
  const float inv = powf(10000.f, -(float)(2 * i) / (float)HDdim);
  const float ang = (float)s * inv;
  const float c = cosf(ang), sn = sinf(ang);
  const long long base = (((long long)s * Bdim + b) * Hdim + h) * HDdim;
  const float t1 = t[base + i], t2 = t[base + halfd + i];
  o[base + i]         = (__bf16)(t1 * c - t2 * sn);
  o[base + halfd + i] = (__bf16)(t2 * c + t1 * sn);
}

// Causal softmax of one score row -> bf16 probs (zeros above diagonal).
__global__ __launch_bounds__(256) void softmax_causal_bf16(
    const float* __restrict__ sc, __bf16* __restrict__ pr, int Sdim, float scale) {
  const int srow = blockIdx.x;
  const long long base = ((long long)blockIdx.y * Sdim + srow) * Sdim;
  const float* row = sc + base;
  __bf16* orow = pr + base;
  __shared__ float red[256];
  float mx = -3.4e38f;
  for (int c = threadIdx.x; c < Sdim; c += 256)
    if (c <= srow) mx = fmaxf(mx, row[c] * scale);
  red[threadIdx.x] = mx;
  __syncthreads();
  for (int o = 128; o > 0; o >>= 1) {
    if (threadIdx.x < o) red[threadIdx.x] = fmaxf(red[threadIdx.x], red[threadIdx.x + o]);
    __syncthreads();
  }
  mx = red[0];
  __syncthreads();
  float sum = 0.f;
  for (int c = threadIdx.x; c < Sdim; c += 256)
    if (c <= srow) sum += expf(row[c] * scale - mx);
  red[threadIdx.x] = sum;
  __syncthreads();
  for (int o = 128; o > 0; o >>= 1) {
    if (threadIdx.x < o) red[threadIdx.x] += red[threadIdx.x + o];
    __syncthreads();
  }
  const float inv = 1.f / red[0];
  for (int c = threadIdx.x; c < Sdim; c += 256)
    orow[c] = (c <= srow) ? (__bf16)(expf(row[c] * scale - mx) * inv) : (__bf16)0.f;
}

// Router: logits = h2 @ router_w, softmax over E=8, top-2, renormalized gates.
__global__ __launch_bounds__(128) void router_gates(
    const __bf16* __restrict__ h2, const float* __restrict__ rw,
    float* __restrict__ gates, int D, int E) {
  const int t = blockIdx.x;
  float acc[E_];
#pragma unroll
  for (int e = 0; e < E_; ++e) acc[e] = 0.f;
  for (int d = threadIdx.x; d < D; d += 128) {
    const float x = (float)h2[(long long)t * D + d];
#pragma unroll
    for (int e = 0; e < E_; ++e) acc[e] += x * rw[(long long)d * E + e];
  }
  __shared__ float red[128][E_];
#pragma unroll
  for (int e = 0; e < E_; ++e) red[threadIdx.x][e] = acc[e];
  __syncthreads();
  for (int o = 64; o > 0; o >>= 1) {
    if (threadIdx.x < o)
#pragma unroll
      for (int e = 0; e < E_; ++e) red[threadIdx.x][e] += red[threadIdx.x + o][e];
    __syncthreads();
  }
  if (threadIdx.x == 0) {
    float lg[E_];
    float mx = -3.4e38f;
#pragma unroll
    for (int e = 0; e < E_; ++e) { lg[e] = red[0][e]; mx = fmaxf(mx, lg[e]); }
    float sum = 0.f;
#pragma unroll
    for (int e = 0; e < E_; ++e) { lg[e] = expf(lg[e] - mx); sum += lg[e]; }
    const float inv = 1.f / sum;
#pragma unroll
    for (int e = 0; e < E_; ++e) lg[e] *= inv;  // probs
    int i1 = 0;
#pragma unroll
    for (int e = 1; e < E_; ++e) if (lg[e] > lg[i1]) i1 = e;
    int i2 = (i1 == 0) ? 1 : 0;
#pragma unroll
    for (int e = 0; e < E_; ++e) if (e != i1 && lg[e] > lg[i2]) i2 = e;
    const float norm = 1.f / (lg[i1] + lg[i2]);
#pragma unroll
    for (int e = 0; e < E_; ++e)
      gates[(long long)t * E + e] = (e == i1) ? lg[i1] * norm
                                   : (e == i2) ? lg[i2] * norm : 0.f;
  }
}

__global__ void silu_mul_bf16(const float* __restrict__ g, const float* __restrict__ u,
                              __bf16* __restrict__ o, long long n) {
  for (long long i = (long long)blockIdx.x * blockDim.x + threadIdx.x; i < n;
       i += (long long)gridDim.x * blockDim.x) {
    const float x = g[i];
    o[i] = (__bf16)((x / (1.f + expf(-x))) * u[i]);
  }
}

// out[s,b,:] = xres[s,b,:] + moe[b*S+s,:]
__global__ void final_merge(const float* __restrict__ xres, const float* __restrict__ moe,
                            float* __restrict__ out, int Sdim, int Bdim, int D) {
  const long long n = (long long)Sdim * Bdim * D;
  for (long long i = (long long)blockIdx.x * blockDim.x + threadIdx.x; i < n;
       i += (long long)gridDim.x * blockDim.x) {
    const int d = (int)(i % D);
    const long long tsb = i / D;
    const int b = (int)(tsb % Bdim);
    const int s = (int)(tsb / Bdim);
    out[i] = xres[i] + moe[((long long)b * Sdim + s) * D + d];
  }
}

// ---------------------------------------------------------------------------
// Host orchestration
// ---------------------------------------------------------------------------
static void launch_gemm(bool transb, const __bf16* A, const __bf16* B, float* Cout,
                        const float* Cin, const float* rowScale, int rsStride,
                        int M, int N, int K, int lda, int ldb, int ldc,
                        long long sA, long long sB, long long sC, int batches,
                        float alpha, hipStream_t st) {
  dim3 g(N / 128, M / 128, batches), b(256);
  if (transb)
    gemm_bf16_wmma<true><<<g, b, 0, st>>>(A, B, Cout, Cin, rowScale, rsStride,
                                          M, N, K, lda, ldb, ldc, sA, sB, sC, alpha);
  else
    gemm_bf16_wmma<false><<<g, b, 0, st>>>(A, B, Cout, Cin, rowScale, rsStride,
                                           M, N, K, lda, ldb, ldc, sA, sB, sC, alpha);
}

static inline int gsz(long long n, int blk) {
  long long g = (n + blk - 1) / blk;
  return (int)(g > 16384 ? 16384 : g);
}

extern "C" void kernel_launch(void* const* d_in, const int* in_sizes, int n_in,
                              void* d_out, int out_size, void* d_ws, size_t ws_size,
                              hipStream_t stream) {
  (void)in_sizes; (void)n_in; (void)out_size; (void)ws_size;
  const float* x     = (const float*)d_in[0];
  const float* n1w   = (const float*)d_in[1];
  const float* n2w   = (const float*)d_in[2];
  const float* Wq    = (const float*)d_in[3];
  const float* Wdkv  = (const float*)d_in[4];
  const float* Wuk   = (const float*)d_in[5];
  const float* Wuv   = (const float*)d_in[6];
  const float* Wo    = (const float*)d_in[7];
  const float* rw    = (const float*)d_in[8];
  const float* Wg    = (const float*)d_in[9];
  const float* Wu    = (const float*)d_in[10];
  const float* Wd    = (const float*)d_in[11];
  float* out = (float*)d_out;

  const int T = S_ * B_;             // 2048 tokens
  size_t off = 0;
  auto wsa = [&](size_t bytes) -> char* {
    char* p = (char*)d_ws + off;
    off += (bytes + 255) & ~(size_t)255;
    return p;
  };

  __bf16* Wq_b   = (__bf16*)wsa((size_t)D_ * H_ * HD_ * 2);
  __bf16* Wdkv_b = (__bf16*)wsa((size_t)D_ * L_ * 2);
  __bf16* Wuk_b  = (__bf16*)wsa((size_t)L_ * H_ * HD_ * 2);
  __bf16* Wuv_b  = (__bf16*)wsa((size_t)L_ * H_ * HD_ * 2);
  __bf16* Wo_b   = (__bf16*)wsa((size_t)H_ * HD_ * D_ * 2);
  __bf16* Wg_b   = (__bf16*)wsa((size_t)E_ * D_ * F_ * 2);
  __bf16* Wu_b   = (__bf16*)wsa((size_t)E_ * D_ * F_ * 2);
  __bf16* Wd_b   = (__bf16*)wsa((size_t)E_ * F_ * D_ * 2);
  __bf16* hb     = (__bf16*)wsa((size_t)T * D_ * 2);
  float*  qf     = (float*)wsa((size_t)T * H_ * HD_ * 4);
  float*  kf     = (float*)wsa((size_t)T * H_ * HD_ * 4);
  float*  vf     = (float*)wsa((size_t)T * H_ * HD_ * 4);
  float*  cf     = (float*)wsa((size_t)T * L_ * 4);
  __bf16* cb     = (__bf16*)wsa((size_t)T * L_ * 2);
  __bf16* qb     = (__bf16*)wsa((size_t)T * H_ * HD_ * 2);
  __bf16* kb     = (__bf16*)wsa((size_t)T * H_ * HD_ * 2);
  __bf16* vb     = (__bf16*)wsa((size_t)T * H_ * HD_ * 2);
  float*  scores = (float*)wsa((size_t)B_ * H_ * S_ * S_ * 4);
  __bf16* probs  = (__bf16*)wsa((size_t)B_ * H_ * S_ * S_ * 2);
  float*  of     = (float*)wsa((size_t)T * H_ * HD_ * 4);
  __bf16* ob     = (__bf16*)wsa((size_t)T * H_ * HD_ * 2);
  float*  xres   = (float*)wsa((size_t)T * D_ * 4);
  __bf16* h2b    = (__bf16*)wsa((size_t)T * D_ * 2);
  float*  gates  = (float*)wsa((size_t)T * E_ * 4);
  float*  gbuf   = (float*)wsa((size_t)T * F_ * 4);
  float*  ubuf   = (float*)wsa((size_t)T * F_ * 4);
  __bf16* actb   = (__bf16*)wsa((size_t)T * F_ * 2);
  float*  moe    = (float*)wsa((size_t)T * D_ * 4);

  auto cast = [&](const float* s, __bf16* dst, long long n) {
    cast_f32_to_bf16<<<gsz(n, 256), 256, 0, stream>>>(s, dst, n);
  };

  // ---- weight bf16 casts ----
  cast(Wq,   Wq_b,   (long long)D_ * H_ * HD_);
  cast(Wdkv, Wdkv_b, (long long)D_ * L_);
  cast(Wuk,  Wuk_b,  (long long)L_ * H_ * HD_);
  cast(Wuv,  Wuv_b,  (long long)L_ * H_ * HD_);
  cast(Wo,   Wo_b,   (long long)H_ * HD_ * D_);
  cast(Wg,   Wg_b,   (long long)E_ * D_ * F_);
  cast(Wu,   Wu_b,   (long long)E_ * D_ * F_);
  cast(Wd,   Wd_b,   (long long)E_ * F_ * D_);

  // ---- h = rmsnorm(x) (token order s*B+b) ----
  rmsnorm_to_bf16<<<T, 256, 0, stream>>>(x, n1w, hb, D_, S_, B_, 0);

  // ---- q = h@Wq ; c = h@Wdkv ; k = c@Wuk ; v = c@Wuv ----
  launch_gemm(false, hb, Wq_b, qf, nullptr, nullptr, 1, T, H_ * HD_, D_,
              D_, H_ * HD_, H_ * HD_, 0, 0, 0, 1, 1.f, stream);
  launch_gemm(false, hb, Wdkv_b, cf, nullptr, nullptr, 1, T, L_, D_,
              D_, L_, L_, 0, 0, 0, 1, 1.f, stream);
  cast(cf, cb, (long long)T * L_);
  launch_gemm(false, cb, Wuk_b, kf, nullptr, nullptr, 1, T, H_ * HD_, L_,
              L_, H_ * HD_, H_ * HD_, 0, 0, 0, 1, 1.f, stream);
  launch_gemm(false, cb, Wuv_b, vf, nullptr, nullptr, 1, T, H_ * HD_, L_,
              L_, H_ * HD_, H_ * HD_, 0, 0, 0, 1, 1.f, stream);

  // ---- RoPE(q), RoPE(k) -> bf16; v -> bf16 ----
  {
    const long long n = (long long)S_ * B_ * H_ * (HD_ / 2);
    rope_to_bf16<<<gsz(n, 256), 256, 0, stream>>>(qf, qb, S_, B_, H_, HD_);
    rope_to_bf16<<<gsz(n, 256), 256, 0, stream>>>(kf, kb, S_, B_, H_, HD_);
  }
  cast(vf, vb, (long long)T * H_ * HD_);

  // ---- scores[b,h] = q[b,h] @ k[b,h]^T  (batched over 64, strided views) ----
  const int ldq = B_ * H_ * HD_;   // 8192
  launch_gemm(true, qb, kb, scores, nullptr, nullptr, 1, S_, S_, HD_,
              ldq, ldq, S_, HD_, HD_, (long long)S_ * S_, B_ * H_, 1.f, stream);

  // ---- causal softmax (scale 1/sqrt(HD)) -> bf16 probs ----
  {
    dim3 g(S_, B_ * H_);
    softmax_causal_bf16<<<g, 256, 0, stream>>>(scores, probs, S_, rsqrtf((float)HD_));
  }

  // ---- o[b,h] = probs[b,h] @ v[b,h]  (batched) ----
  launch_gemm(false, probs, vb, of, nullptr, nullptr, 1, S_, HD_, S_,
              S_, ldq, ldq, (long long)S_ * S_, HD_, HD_, B_ * H_, 1.f, stream);

  // ---- xres = x + o @ Wo ----
  cast(of, ob, (long long)T * H_ * HD_);
  launch_gemm(false, ob, Wo_b, xres, x, nullptr, 1, T, D_, H_ * HD_,
              H_ * HD_, D_, D_, 0, 0, 0, 1, 1.f, stream);

  // ---- h2 = rmsnorm(xres) in t = b*S+s order ----
  rmsnorm_to_bf16<<<T, 256, 0, stream>>>(xres, n2w, h2b, D_, S_, B_, 1);

  // ---- router gates (softmax + top-2, renormalized) ----
  router_gates<<<T, 128, 0, stream>>>(h2b, rw, gates, D_, E_);

  // ---- dense MoE (matches reference: every expert over every token) ----
  zero_f32<<<gsz((long long)T * D_, 256), 256, 0, stream>>>(moe, (long long)T * D_);
  for (int e = 0; e < E_; ++e) {
    const __bf16* wg = Wg_b + (long long)e * D_ * F_;
    const __bf16* wu = Wu_b + (long long)e * D_ * F_;
    const __bf16* wd = Wd_b + (long long)e * F_ * D_;
    launch_gemm(false, h2b, wg, gbuf, nullptr, nullptr, 1, T, F_, D_,
                D_, F_, F_, 0, 0, 0, 1, 1.f, stream);
    launch_gemm(false, h2b, wu, ubuf, nullptr, nullptr, 1, T, F_, D_,
                D_, F_, F_, 0, 0, 0, 1, 1.f, stream);
    silu_mul_bf16<<<gsz((long long)T * F_, 256), 256, 0, stream>>>(
        gbuf, ubuf, actb, (long long)T * F_);
    // moe += gates[:,e] * (act @ Wd[e])
    launch_gemm(false, actb, wd, moe, moe, gates + e, E_, T, D_, F_,
                F_, D_, D_, 0, 0, 0, 1, 1.f, stream);
  }

  // ---- out = xres + moe (transposed back to (S,B,D)) ----
  final_merge<<<gsz((long long)T * D_, 256), 256, 0, stream>>>(xres, moe, out, S_, B_, D_);
}
